// Decoder_21981642621050
// MI455X (gfx1250) — matrix operand order repaired and checked by reading, not compiled, not run on hardware
//
#include <hip/hip_runtime.h>
#include <stdint.h>

// ---------------- problem constants ----------------
#define BB   256
#define TT   25
#define NN   21
#define F_IN 8
#define HID  512
#define LAT  32
#define ENC  96
#define OUT_ 64
#define HTS  40        // padded row stride (ushorts) for transposed h/yi buffers

#define OFF_LOCD  (BB*TT*NN*4)          // 537600
#define OFF_LOGZ  (2*BB*TT*NN*4)        // 1075200

// ---------------- WMMA types ----------------
typedef __attribute__((ext_vector_type(16))) __bf16 v16bf;
typedef __attribute__((ext_vector_type(8)))  float  v8f;
typedef __attribute__((ext_vector_type(8)))  unsigned int v8u;

__device__ __forceinline__ v16bf as_v16bf(v8u u) { return __builtin_bit_cast(v16bf, u); }

__device__ __forceinline__ unsigned short f2bf(float f) {
    unsigned int u = __float_as_uint(f);
    unsigned int r = u + 0x7FFFu + ((u >> 16) & 1u);   // RNE
    return (unsigned short)(r >> 16);
}
__device__ __forceinline__ unsigned int pk2bf(float lo, float hi) {
    return (unsigned int)f2bf(lo) | ((unsigned int)f2bf(hi) << 16);
}

// fast transcendentals: v_exp_f32 + v_rcp_f32 (approximation error << bf16 noise)
__device__ __forceinline__ float fast_rcp(float x) { return __builtin_amdgcn_rcpf(x); }
__device__ __forceinline__ float sigm(float x) { return fast_rcp(1.0f + __expf(-x)); }
__device__ __forceinline__ float fast_tanh(float x) {
    return 1.0f - 2.0f * fast_rcp(__expf(2.0f * x) + 1.0f);
}

__device__ __forceinline__ v8f wmma_bf16(v16bf a, v16bf b, v8f c) {
    return __builtin_amdgcn_wmma_f32_16x16x32_bf16(false, a, false, b, (short)0, c, false, false);
}

// A fragment (16x32 bf16) from a row-major bf16 row: two 16B LDS loads.
__device__ __forceinline__ v16bf ldsA(const unsigned short* rowptr, int k0, int half) {
    const uint4* p = reinterpret_cast<const uint4*>(rowptr + k0 + half * 16);
    uint4 lo = p[0];    // K = k0 + half*16 + {0..7}
    uint4 hi = p[1];    // K = k0 + half*16 + 8 + {0..7}
    v8u u;
    u[0] = lo.x; u[1] = lo.y; u[2] = lo.z; u[3] = lo.w;
    u[4] = hi.x; u[5] = hi.y; u[6] = hi.z; u[7] = hi.w;
    return as_v16bf(u);
}

// B fragment from a transposed bf16 column (row of hT): K = half*16 + {0..15} contiguous.
__device__ __forceinline__ v16bf ldBT(const unsigned short* colptr, int half) {
    const uint4* p = reinterpret_cast<const uint4*>(colptr + half * 16);
    uint4 a = p[0], b = p[1];
    v8u u;
    u[0] = a.x; u[1] = a.y; u[2] = a.z; u[3] = a.w;
    u[4] = b.x; u[5] = b.y; u[6] = b.z; u[7] = b.w;
    return as_v16bf(u);
}

// B fragment: 8 contiguous dwords per lane from a packed weight buffer
__device__ __forceinline__ v16bf ldB(const unsigned int* p) {
    const uint4* q = reinterpret_cast<const uint4*>(p);
    uint4 a = q[0], b = q[1];
    v8u u;
    u[0] = a.x; u[1] = a.y; u[2] = a.z; u[3] = a.w;
    u[4] = b.x; u[5] = b.y; u[6] = b.z; u[7] = b.w;
    return as_v16bf(u);
}

// ---------------- weight pre-pack kernels ----------------
// B-matrix lane mapping for one 32x16 tile: n = nt*16 + lane%16 ; K = (lane/16)*16 + 2j {,+1}.
__global__ __launch_bounds__(32)
void pack_gate_kernel(const float* __restrict__ Wi, const float* __restrict__ Wh,
                      unsigned int* __restrict__ pk) {
    int tile = blockIdx.x;            // 0 .. 17*128-1
    int kt = tile >> 7;
    int nt = tile & 127;
    int lane = threadIdx.x;
    int half = lane >> 4, nl = lane & 15;
    int n = nt * 16 + nl;
    unsigned int* dst = pk + ((size_t)tile * 32 + lane) * 8;
#pragma unroll
    for (int j = 0; j < 8; ++j) {
        int kb = half * 16 + 2 * j;
        float f0, f1;
        if (kt < 16) {
            int k = kt * 32 + kb;
            f0 = Wh[(size_t)k * 2048 + n];
            f1 = Wh[(size_t)(k + 1) * 2048 + n];
        } else {
            f0 = (kb     < F_IN) ? Wi[(size_t)kb       * 2048 + n] : 0.0f;
            f1 = (kb + 1 < F_IN) ? Wi[(size_t)(kb + 1) * 2048 + n] : 0.0f;
        }
        dst[j] = pk2bf(f0, f1);
    }
}

__global__ __launch_bounds__(32)
void pack_fc_kernel(const float* __restrict__ Wfc, const float* __restrict__ Wfc2,
                    unsigned int* __restrict__ pk1, unsigned int* __restrict__ pk2) {
    int blk = blockIdx.x;             // 0..127 ; tile = kt*4+nt
    int mat = blk >> 6;
    int r = blk & 63;
    const float* src = mat ? Wfc2 : Wfc;
    unsigned int* dst = (mat ? pk2 : pk1) + ((size_t)r * 32 + threadIdx.x) * 8;
    int half = threadIdx.x >> 4, nl = threadIdx.x & 15;
    int kt = r >> 2, nt = r & 3;
    int n = nt * 16 + nl;
#pragma unroll
    for (int j = 0; j < 8; ++j) {
        int k = kt * 32 + half * 16 + 2 * j;
        dst[j] = pk2bf(src[(size_t)k * 64 + n], src[(size_t)(k + 1) * 64 + n]);
    }
}

__global__ __launch_bounds__(32)
void pack_init_kernel(const float* __restrict__ Wh1, const float* __restrict__ Wh2,
                      unsigned int* __restrict__ pk1, unsigned int* __restrict__ pk2) {
    int blk = blockIdx.x;             // 0..255 ; tile = kt*32+nt
    int mat = blk >> 7;
    int r = blk & 127;
    const float* src = mat ? Wh2 : Wh1;
    unsigned int* dst = (mat ? pk2 : pk1) + ((size_t)r * 32 + threadIdx.x) * 8;
    int half = threadIdx.x >> 4, nl = threadIdx.x & 15;
    int kt = r >> 5, nt = r & 31;
    int n = nt * 16 + nl;
#pragma unroll
    for (int j = 0; j < 8; ++j) {
        int k = kt * 32 + half * 16 + 2 * j;
        dst[j] = pk2bf(src[(size_t)k * HID + n], src[(size_t)(k + 1) * HID + n]);
    }
}

// ---------------- LDS layout (dynamic shared) ----------------
constexpr int ALGN16(int x) { return (x + 15) & ~15; }
constexpr int OFFS_G    = 0;                                    // float[441]
constexpr int OFFS_GROW = ALGN16(OFFS_G    + 441 * 4);          // float[21]
constexpr int OFFS_C    = ALGN16(OFFS_GROW + 21 * 4);           // float[21*512]
constexpr int OFFS_HT   = ALGN16(OFFS_C    + 21 * 512 * 4);     // ushort[512*HTS] bf16, transposed h
constexpr int OFFS_YIT  = ALGN16(OFFS_HT   + 512 * HTS * 2);    // ushort[512*HTS] bf16, transposed yi
constexpr int OFFS_GM   = ALGN16(OFFS_YIT  + 512 * HTS * 2);    // ushort[32*512] bf16
constexpr int OFFS_GXI  = ALGN16(OFFS_GM   + 32 * 512 * 2);     // ushort[32*32]
constexpr int OFFS_GBF  = ALGN16(OFFS_GXI  + 32 * 32 * 2);      // ushort[32*32] bf16 G
constexpr int OFFS_YI1  = ALGN16(OFFS_GBF  + 32 * 32 * 2);      // float[21*64]
constexpr int OFFS_YI2  = ALGN16(OFFS_YI1  + 21 * 64 * 4);
constexpr int OFFS_WLOC = ALGN16(OFFS_YI2  + 21 * 64 * 4);      // float[192]
constexpr int OFFS_WZ   = ALGN16(OFFS_WLOC + 192 * 4);          // float[192]
constexpr int OFFS_P    = ALGN16(OFFS_WZ   + 192 * 4);          // float[2*21*3] (yi1@Wloc, yi2@Wz)
constexpr int OFFS_LST  = ALGN16(OFFS_P    + 126 * 4);          // float[21*4]
constexpr int OFFS_XI   = ALGN16(OFFS_LST  + 21 * 4 * 4);       // float[21*8]
constexpr int SMEM_BYTES = ALGN16(OFFS_XI  + 21 * 8 * 4);       // ~173 KB

// ---------------- main persistent decoder kernel: one workgroup per batch element ----------------
__global__ __launch_bounds__(256)
void decoder_kernel(const float* __restrict__ xg,   const float* __restrict__ encg,
                    const float* __restrict__ zg,   const float* __restrict__ Gg,
                    const float* __restrict__ b_lstm,
                    const float* __restrict__ bh1,  const float* __restrict__ bh2,
                    const float* __restrict__ bfc,  const float* __restrict__ bfc2,
                    const float* __restrict__ Wloc, const float* __restrict__ bloc,
                    const float* __restrict__ Wz,   const float* __restrict__ bz,
                    const unsigned int* __restrict__ pkGate,
                    const unsigned int* __restrict__ pkFc,
                    const unsigned int* __restrict__ pkFc2,
                    const unsigned int* __restrict__ pkWh1,
                    const unsigned int* __restrict__ pkWh2,
                    float* __restrict__ out) {
    extern __shared__ char smem[];
    float*          G_s    = (float*)(smem + OFFS_G);
    float*          Grow_s = (float*)(smem + OFFS_GROW);
    float*          c_s    = (float*)(smem + OFFS_C);
    unsigned short* hT_s   = (unsigned short*)(smem + OFFS_HT);
    unsigned short* yiT_s  = (unsigned short*)(smem + OFFS_YIT);
    unsigned short* Gm_s   = (unsigned short*)(smem + OFFS_GM);
    unsigned short* Gxi_s  = (unsigned short*)(smem + OFFS_GXI);
    unsigned short* Gbf_s  = (unsigned short*)(smem + OFFS_GBF);
    float*          yi1_s  = (float*)(smem + OFFS_YI1);
    float*          yi2_s  = (float*)(smem + OFFS_YI2);
    float*          Wloc_s = (float*)(smem + OFFS_WLOC);
    float*          Wz_s   = (float*)(smem + OFFS_WZ);
    float*          P_s    = (float*)(smem + OFFS_P);
    float*          lst_s  = (float*)(smem + OFFS_LST);
    float*          xi_s   = (float*)(smem + OFFS_XI);

    const int tid  = threadIdx.x;
    const int b    = blockIdx.x;
    const int wave = tid >> 5;
    const int lane = tid & 31;
    const int half = lane >> 4;
    const int nl   = lane & 15;
    const v8f vzero = {0.f,0.f,0.f,0.f,0.f,0.f,0.f,0.f};

    // ---------- init loads + zeroing ----------
    for (int i = tid; i < 441; i += 256) G_s[i] = Gg[i];
    {
        unsigned int* p;
        p = (unsigned int*)Gm_s;  for (int i = tid; i < 32 * 512 / 2;  i += 256) p[i] = 0u;
        p = (unsigned int*)Gxi_s; for (int i = tid; i < 32 * 32 / 2;   i += 256) p[i] = 0u;
        p = (unsigned int*)Gbf_s; for (int i = tid; i < 32 * 32 / 2;   i += 256) p[i] = 0u;
        p = (unsigned int*)hT_s;  for (int i = tid; i < 512 * HTS / 2; i += 256) p[i] = 0u;
        p = (unsigned int*)yiT_s; for (int i = tid; i < 512 * HTS / 2; i += 256) p[i] = 0u;
    }
    if (tid < NN * F_IN) xi_s[tid] = xg[(size_t)b * NN * F_IN + tid];
    if (tid < NN * 4)    lst_s[tid] = xg[(size_t)b * NN * F_IN + (tid >> 2) * F_IN + (tid & 3)];
    if (tid < 192) { Wloc_s[tid] = Wloc[tid]; Wz_s[tid] = Wz[tid]; }
    __syncthreads();

    // Grow, bf16 G, Genc (= bf16(G @ [z|enc]) into Gm rows, k<128)
    if (tid < NN) {
        float s = 0.f;
        for (int m = 0; m < NN; ++m) s += G_s[tid * NN + m];
        Grow_s[tid] = s;
    }
    if (tid < NN * NN) {
        int n = tid / NN, m = tid % NN;
        Gbf_s[n * 32 + m] = f2bf(G_s[tid]);
    }
    for (int i = tid; i < NN * 32; i += 256) {       // 4-wide over k
        int n = i >> 5, k4 = (i & 31) << 2;
        float s0 = 0.f, s1 = 0.f, s2 = 0.f, s3 = 0.f;
        for (int m = 0; m < NN; ++m) {
            float g = G_s[n * NN + m];
            float4 v;
            if (k4 < LAT) v = *(const float4*)(zg + (size_t)b * NN * LAT + m * LAT + k4);
            else          v = *(const float4*)(encg + (size_t)b * NN * ENC + m * ENC + (k4 - LAT));
            s0 += g * v.x; s1 += g * v.y; s2 += g * v.z; s3 += g * v.w;
        }
        uint2 o; o.x = pk2bf(s0, s1); o.y = pk2bf(s2, s3);
        *(uint2*)(Gm_s + n * HID + k4) = o;
    }
    __syncthreads();

    // h0/c0 via WMMA: Genc(21x128) @ Wh1/Wh2 (128x512)
    {
        const int mat = wave >> 2;
        const unsigned int* pkW = mat ? pkWh2 : pkWh1;
        const float* bias = mat ? bh2 : bh1;
        for (int nn2 = 0; nn2 < 8; ++nn2) {
            const int nt = (wave & 3) * 8 + nn2;
            const unsigned int* pB = pkW + ((size_t)nt * 32 + lane) * 8;
            v8f c0 = vzero, c1 = vzero;
#pragma unroll
            for (int kt = 0; kt < 4; ++kt) {
                v16bf a0 = ldsA(Gm_s + nl * HID,        kt * 32, half);
                v16bf a1 = ldsA(Gm_s + (16 + nl) * HID, kt * 32, half);
                v16bf bf = ldB(pB + kt * 8192);
                c0 = wmma_bf16(a0, bf, c0);
                c1 = wmma_bf16(a1, bf, c1);
            }
            const int col = nt * 16 + nl;
            const float bv = bias[col];
            if (mat == 0) {                        // wave-uniform branch
#pragma unroll
                for (int r = 0; r < 8; ++r) {
                    int m = r + half * 8;
                    hT_s[col * HTS + m] = f2bf(c0[r] + Grow_s[m] * bv);
                }
                if (half == 0) {
#pragma unroll
                    for (int r = 0; r < 5; ++r) {
                        int m = 16 + r;
                        hT_s[col * HTS + m] = f2bf(c1[r] + Grow_s[m] * bv);
                    }
                }
            } else {
#pragma unroll
                for (int r = 0; r < 8; ++r) {
                    int m = r + half * 8;
                    c_s[m * HID + col] = c0[r] + Grow_s[m] * bv;
                }
                if (half == 0) {
#pragma unroll
                    for (int r = 0; r < 5; ++r) {
                        int m = 16 + r;
                        c_s[m * HID + col] = c1[r] + Grow_s[m] * bv;
                    }
                }
            }
        }
    }
    __syncthreads();

    // hoisted per-lane constants for the step loop
    float blr[4][4];
#pragma unroll
    for (int tt2 = 0; tt2 < 4; ++tt2) {
        int uu = (wave * 4 + tt2) * 16 + nl;
        blr[tt2][0] = b_lstm[uu];
        blr[tt2][1] = b_lstm[512 + uu];
        blr[tt2][2] = b_lstm[1024 + uu];
        blr[tt2][3] = b_lstm[1536 + uu];
    }
    const int matD = wave >> 2;
    const int ntD  = wave & 3;
    const unsigned int* pkWD = matD ? pkFc2 : pkFc;
    const int colD = ntD * 16 + nl;
    const float biasD = (matD ? bfc2 : bfc)[colD];
    float bl0 = 0.f, bl1 = 0.f, bl2 = 0.f, bz0 = 0.f, bz1 = 0.f, bz2 = 0.f, grn = 0.f;
    if (tid < NN) {
        grn = Grow_s[tid];
        bl0 = bloc[0]; bl1 = bloc[1]; bl2 = bloc[2];
        bz0 = bz[0];   bz1 = bz[1];   bz2 = bz[2];
    }

    // ---------- T recurrent steps ----------
    for (int step = 0; step < TT; ++step) {
        // Stage A: Gm = bf16(G @ h) via WMMA; Gxi in VALU
        {
            v16bf ga0 = ldsA(Gbf_s + nl * 32,        0, half);
            v16bf ga1 = ldsA(Gbf_s + (16 + nl) * 32, 0, half);
#pragma unroll
            for (int q = 0; q < 4; ++q) {
                const int col = (wave * 4 + q) * 16 + nl;
                v16bf bh = ldBT(hT_s + col * HTS, half);
                v8f d0 = wmma_bf16(ga0, bh, vzero);
                v8f d1 = wmma_bf16(ga1, bh, vzero);
#pragma unroll
                for (int r = 0; r < 8; ++r) Gm_s[(r + half * 8) * HID + col] = f2bf(d0[r]);
                if (half == 0) {
#pragma unroll
                    for (int r = 0; r < 5; ++r) Gm_s[(16 + r) * HID + col] = f2bf(d1[r]);
                }
            }
        }
        if (tid < NN * F_IN) {
            int n = tid >> 3, k = tid & 7;
            float s = 0.f;
            for (int m = 0; m < NN; ++m) s += G_s[n * NN + m] * xi_s[m * F_IN + k];
            Gxi_s[n * 32 + k] = f2bf(s);
        }
        __syncthreads();

        // Stage B: gates GEMM (WMMA) + LSTM pointwise straight from accumulators
        for (int ttile = 0; ttile < 4; ++ttile) {
            const int t = wave * 4 + ttile;               // hidden tile 0..31
            const int u = t * 16 + nl;
            const unsigned int* pB = pkGate + ((size_t)t * 32 + lane) * 8;
            v8f acc[4][2];
#pragma unroll
            for (int g = 0; g < 4; ++g) { acc[g][0] = vzero; acc[g][1] = vzero; }

            for (int kt = 0; kt < 16; ++kt) {
                v16bf a0 = ldsA(Gm_s + nl * HID,        kt * 32, half);
                v16bf a1 = ldsA(Gm_s + (16 + nl) * HID, kt * 32, half);
                v16bf bfr[4];
#pragma unroll
                for (int g = 0; g < 4; ++g) bfr[g] = ldB(pB + g * 8192);
#pragma unroll
                for (int g = 0; g < 4; ++g) {
                    acc[g][0] = wmma_bf16(a0, bfr[g], acc[g][0]);
                    acc[g][1] = wmma_bf16(a1, bfr[g], acc[g][1]);
                }
                pB += 32768;
            }
            {   // xi K-tile (kt == 16)
                v16bf a0 = ldsA(Gxi_s + nl * 32,        0, half);
                v16bf a1 = ldsA(Gxi_s + (16 + nl) * 32, 0, half);
                v16bf bfr[4];
#pragma unroll
                for (int g = 0; g < 4; ++g) bfr[g] = ldB(pB + g * 8192);
#pragma unroll
                for (int g = 0; g < 4; ++g) {
                    acc[g][0] = wmma_bf16(a0, bfr[g], acc[g][0]);
                    acc[g][1] = wmma_bf16(a1, bfr[g], acc[g][1]);
                }
            }

            // LSTM pointwise; packed vector stores into hT/yiT
            unsigned int hpk[4], ypk[4];
#pragma unroll
            for (int rp = 0; rp < 4; ++rp) {
                float hnv[2], ynv[2];
#pragma unroll
                for (int s2 = 0; s2 < 2; ++s2) {
                    int r = rp * 2 + s2;
                    int m = r + half * 8;
                    float gr = Grow_s[m];
                    float ip = acc[0][0][r] + gr * blr[ttile][0];
                    float fp = acc[1][0][r] + gr * blr[ttile][1];
                    float gp = acc[2][0][r] + gr * blr[ttile][2];
                    float op = acc[3][0][r] + gr * blr[ttile][3];
                    float co = c_s[m * HID + u];
                    float cn = sigm(fp) * co + sigm(ip) * fast_tanh(gp);
                    float hn = sigm(op) * fast_tanh(cn);
                    c_s[m * HID + u] = cn;
                    hnv[s2] = hn;
                    ynv[s2] = fast_tanh(hn);
                }
                hpk[rp] = pk2bf(hnv[0], hnv[1]);
                ypk[rp] = pk2bf(ynv[0], ynv[1]);
            }
            {
                uint4 v;
                v.x = hpk[0]; v.y = hpk[1]; v.z = hpk[2]; v.w = hpk[3];
                *(uint4*)(hT_s + u * HTS + 8 * half) = v;
                v.x = ypk[0]; v.y = ypk[1]; v.z = ypk[2]; v.w = ypk[3];
                *(uint4*)(yiT_s + u * HTS + 8 * half) = v;
            }
            if (half == 0) {
                float hnv[5], ynv[5];
#pragma unroll
                for (int r = 0; r < 5; ++r) {
                    int m = 16 + r;
                    float gr = Grow_s[m];
                    float ip = acc[0][1][r] + gr * blr[ttile][0];
                    float fp = acc[1][1][r] + gr * blr[ttile][1];
                    float gp = acc[2][1][r] + gr * blr[ttile][2];
                    float op = acc[3][1][r] + gr * blr[ttile][3];
                    float co = c_s[m * HID + u];
                    float cn = sigm(fp) * co + sigm(ip) * fast_tanh(gp);
                    float hn = sigm(op) * fast_tanh(cn);
                    c_s[m * HID + u] = cn;
                    hnv[r] = hn;
                    ynv[r] = fast_tanh(hn);
                }
                uint2 h2; h2.x = pk2bf(hnv[0], hnv[1]); h2.y = pk2bf(hnv[2], hnv[3]);
                *(uint2*)(hT_s + u * HTS + 16) = h2;
                hT_s[u * HTS + 20] = f2bf(hnv[4]);
                uint2 y2; y2.x = pk2bf(ynv[0], ynv[1]); y2.y = pk2bf(ynv[2], ynv[3]);
                *(uint2*)(yiT_s + u * HTS + 16) = y2;
                yiT_s[u * HTS + 20] = f2bf(ynv[4]);
            }
        }
        __syncthreads();

        // Stage C: Gm = bf16(G @ yi) via WMMA
        {
            v16bf ga0 = ldsA(Gbf_s + nl * 32,        0, half);
            v16bf ga1 = ldsA(Gbf_s + (16 + nl) * 32, 0, half);
#pragma unroll
            for (int q = 0; q < 4; ++q) {
                const int col = (wave * 4 + q) * 16 + nl;
                v16bf by = ldBT(yiT_s + col * HTS, half);
                v8f d0 = wmma_bf16(ga0, by, vzero);
                v8f d1 = wmma_bf16(ga1, by, vzero);
#pragma unroll
                for (int r = 0; r < 8; ++r) Gm_s[(r + half * 8) * HID + col] = f2bf(d0[r]);
                if (half == 0) {
#pragma unroll
                    for (int r = 0; r < 5; ++r) Gm_s[(16 + r) * HID + col] = f2bf(d1[r]);
                }
            }
        }
        __syncthreads();

        // Stage D: yi1/yi2 = tanh(Gyi @ Wfc{,2} + Grow*b) — one (mat, ntile) per wave
        {
            v8f c0 = vzero, c1 = vzero;
            const unsigned int* pD = pkWD + ((size_t)ntD * 32 + lane) * 8;
            for (int kt = 0; kt < 16; ++kt) {
                v16bf a0 = ldsA(Gm_s + nl * HID,        kt * 32, half);
                v16bf a1 = ldsA(Gm_s + (16 + nl) * HID, kt * 32, half);
                v16bf bf = ldB(pD);
                c0 = wmma_bf16(a0, bf, c0);
                c1 = wmma_bf16(a1, bf, c1);
                pD += 1024;
            }
            float* dst = matD ? yi2_s : yi1_s;
#pragma unroll
            for (int r = 0; r < 8; ++r) {
                int m = r + half * 8;
                dst[m * OUT_ + colD] = fast_tanh(c0[r] + Grow_s[m] * biasD);
            }
            if (half == 0) {
#pragma unroll
                for (int r = 0; r < 5; ++r) {
                    int m = 16 + r;
                    dst[m * OUT_ + colD] = fast_tanh(c1[r] + Grow_s[m] * biasD);
                }
            }
        }
        __syncthreads();

        // Stage E': project first — P1 = yi1@Wloc, P2 = yi2@Wz   (21x3 each)
        if (tid < 2 * NN * 3) {
            int sel = tid / (NN * 3);
            int id2 = tid % (NN * 3);
            int m = id2 / 3, j = id2 % 3;
            const float* src = sel ? yi2_s : yi1_s;
            const float* W   = sel ? Wz_s : Wloc_s;
            float s = 0.f;
            for (int k = 0; k < OUT_; ++k) s += src[m * OUT_ + k] * W[k * 3 + j];
            P_s[tid] = s;     // P_s[sel*63 + m*3 + j]
        }
        __syncthreads();

        // Stage G: G-mix of P (tiny), quaternion normalize + Hamilton, outputs, carries
        if (tid < NN) {
            const int n = tid;
            float l0 = grn * bl0, l1 = grn * bl1, l2 = grn * bl2;
            float z0 = grn * bz0, z1 = grn * bz1, z2 = grn * bz2;
            for (int m = 0; m < NN; ++m) {
                float g = G_s[n * NN + m];
                l0 += g * P_s[m * 3 + 0];
                l1 += g * P_s[m * 3 + 1];
                l2 += g * P_s[m * 3 + 2];
                z0 += g * P_s[63 + m * 3 + 0];
                z1 += g * P_s[63 + m * 3 + 1];
                z2 += g * P_s[63 + m * 3 + 2];
            }
            float inv = 1.0f / sqrtf(1.0f + l0 * l0 + l1 * l1 + l2 * l2);
            float dw = inv, dx = l0 * inv, dy = l1 * inv, dz = l2 * inv;
            float qw = lst_s[n * 4 + 0], qx = lst_s[n * 4 + 1];
            float qy = lst_s[n * 4 + 2], qz = lst_s[n * 4 + 3];
            float lw = qw * dw - qx * dx - qy * dy - qz * dz;
            float lx = qw * dx + qx * dw + qy * dz - qz * dy;
            float ly = qw * dy - qx * dz + qy * dw + qz * dx;
            float lz = qw * dz + qx * dy - qy * dx + qz * dw;
            size_t b4 = (((size_t)b * TT + step) * NN + n) * 4;
            out[b4 + 0] = lw; out[b4 + 1] = lx; out[b4 + 2] = ly; out[b4 + 3] = lz;
            out[OFF_LOCD + b4 + 0] = dw; out[OFF_LOCD + b4 + 1] = dx;
            out[OFF_LOCD + b4 + 2] = dy; out[OFF_LOCD + b4 + 3] = dz;
            size_t b3 = (((size_t)b * TT + step) * NN + n) * 3;
            out[OFF_LOGZ + b3 + 0] = z0;
            out[OFF_LOGZ + b3 + 1] = z1;
            out[OFF_LOGZ + b3 + 2] = z2;
            lst_s[n * 4 + 0] = lw; lst_s[n * 4 + 1] = lx;
            lst_s[n * 4 + 2] = ly; lst_s[n * 4 + 3] = lz;
            xi_s[n * 8 + 0] = lw; xi_s[n * 8 + 1] = lx;
            xi_s[n * 8 + 2] = ly; xi_s[n * 8 + 3] = lz;
            xi_s[n * 8 + 4] = dw; xi_s[n * 8 + 5] = dx;
            xi_s[n * 8 + 6] = dy; xi_s[n * 8 + 7] = dz;
        }
        __syncthreads();
    }
}

// ---------------- host-side launcher ----------------
extern "C" void kernel_launch(void* const* d_in, const int* in_sizes, int n_in,
                              void* d_out, int out_size, void* d_ws, size_t ws_size,
                              hipStream_t stream) {
    (void)in_sizes; (void)n_in; (void)out_size; (void)ws_size;
    const float* x      = (const float*)d_in[0];
    const float* enc    = (const float*)d_in[1];
    const float* z      = (const float*)d_in[2];
    /* d_in[3] = y (unused by reference) */
    const float* G      = (const float*)d_in[4];
    const float* Wi     = (const float*)d_in[5];
    const float* Wh     = (const float*)d_in[6];
    const float* b_lstm = (const float*)d_in[7];
    const float* Wh1    = (const float*)d_in[8];
    const float* bh1    = (const float*)d_in[9];
    const float* Wh2    = (const float*)d_in[10];
    const float* bh2    = (const float*)d_in[11];
    const float* Wfc    = (const float*)d_in[12];
    const float* bfc    = (const float*)d_in[13];
    const float* Wfc2   = (const float*)d_in[14];
    const float* bfc2   = (const float*)d_in[15];
    const float* Wloc   = (const float*)d_in[16];
    const float* bloc   = (const float*)d_in[17];
    const float* Wz     = (const float*)d_in[18];
    const float* bz     = (const float*)d_in[19];

    unsigned int* pkGate = (unsigned int*)d_ws;                 // 17*128*256 dw
    unsigned int* pkFc   = pkGate + (size_t)17 * 128 * 256;     // 16*4*256 dw
    unsigned int* pkFc2  = pkFc   + (size_t)64 * 256;
    unsigned int* pkWh1  = pkFc2  + (size_t)64 * 256;           // 4*32*256 dw
    unsigned int* pkWh2  = pkWh1  + (size_t)128 * 256;

    pack_gate_kernel<<<17 * 128, 32, 0, stream>>>(Wi, Wh, pkGate);
    pack_fc_kernel<<<128, 32, 0, stream>>>(Wfc, Wfc2, pkFc, pkFc2);
    pack_init_kernel<<<256, 32, 0, stream>>>(Wh1, Wh2, pkWh1, pkWh2);

    decoder_kernel<<<BB, 256, SMEM_BYTES, stream>>>(
        x, enc, z, G, b_lstm, bh1, bh2,
        bfc, bfc2, Wloc, bloc, Wz, bz,
        pkGate, pkFc, pkFc2, pkWh1, pkWh2, (float*)d_out);
}
